// DARTSModelLayers_16990890623330
// MI455X (gfx1250) — compile-verified, matrix operand order checked by reading, not run on hardware
//
#include <hip/hip_runtime.h>
#include <math.h>

// ---------------------------------------------------------------------------
// Problem dims (fixed by reference)
// ---------------------------------------------------------------------------
#define B_DIM 4096
#define N_DIM 512
#define H_DIM 1024
#define T_ITERS 10

typedef __attribute__((ext_vector_type(16))) __bf16 v16bf;
typedef __attribute__((ext_vector_type(4)))  __bf16 v4bf;
typedef __attribute__((ext_vector_type(8)))  float  v8f;

static __device__ inline __bf16 f2bf(float f) { return (__bf16)f; }

static __device__ inline v8f wmma_bf16(v16bf a, v16bf b, v8f c) {
  // D = A(16x32 bf16) x B(32x16 bf16) + C(f32)
  return __builtin_amdgcn_wmma_f32_16x16x32_bf16(
      /*neg_a=*/false, a, /*neg_b=*/false, b,
      /*c_mod=*/(short)0, c, /*reuse_a=*/false, /*reuse_b=*/false);
}

// ---------------------------------------------------------------------------
// Workspace layout (bytes, 256-aligned)
// ---------------------------------------------------------------------------
#define OFF_SCAL   ((size_t)0)               // [0]=L [1]=1/L [2]=0.001/L
#define OFF_AW     ((size_t)256)             // T*20 f32 softmax(alpha)
#define OFF_BW     ((size_t)1280)            // T*2  f32 softmax(beta)
#define OFF_WBF    ((size_t)1536)            // [N,H] bf16   (1 MB)
#define OFF_WTBF   ((size_t)1050112)         // [H,N] bf16   (1 MB)
#define OFF_ZAUX   ((size_t)2098688)         // [B,H] bf16   (8 MB)
#define OFF_RBF    ((size_t)10487296)        // [B,N] bf16   (4 MB)
#define OFF_ZP     ((size_t)14681600)        // [B,H] f32    (16 MB)

// ---------------------------------------------------------------------------
// 20-way activation mixture (matches reference op list / order)
// ---------------------------------------------------------------------------
static __device__ inline float mix20(float x, const float aw[20], float lam) {
  float ax   = fabsf(x);
  float em   = __expf(-ax);                       // e^{-|x|} in (0,1]
  float l1p  = log1pf(em);
  float sp   = fmaxf(x, 0.f) + l1p;               // softplus (stable)
  float lsig = fminf(x, 0.f) - l1p;               // log_sigmoid (stable)
  float sig  = (x >= 0.f) ? 1.f / (1.f + em) : em / (1.f + em);
  float th   = tanhf(x);
  float exm1 = (x > 0.f) ? x : (__expf(x) - 1.f); // elu / celu(alpha=1)
  float r6   = fminf(fmaxf(x + 3.f, 0.f), 6.f);   // relu6(x+3)

  float r = 0.f;
  r += aw[0]  * ((x > lam) ? (x - lam) : ((x < -lam) ? (x + lam) : 0.f)); // softshrink
  r += aw[1]  * fmaxf(x, 0.f);                                            // relu
  r += aw[2]  * x;                                                        // identity
  r += aw[3]  * (0.5f * x * (1.f + erff(x * 0.70710678118654752f)));      // gelu exact
  r += aw[4]  * exm1;                                                     // elu
  r += aw[5]  * ((ax > lam) ? x : 0.f);                                   // hardshrink
  r += aw[6]  * fminf(fmaxf(x, -1.f), 1.f);                               // hardtanh
  r += aw[7]  * (x * r6 * (1.f / 6.f));                                   // hardswish
  r += aw[8]  * (1.0507009873554805f *
                 ((x > 0.f) ? x : (1.6732632423543772f * (__expf(x) - 1.f)))); // selu
  r += aw[9]  * exm1;                                                     // celu(1.0)
  r += aw[10] * ((x > 0.f) ? x : 0.01f * x);                              // leaky_relu
  r += aw[11] * lsig;                                                     // log_sigmoid
  r += aw[12] * (x - th);                                                 // tanhshrink
  r += aw[13] * (x / (1.f + ax));                                         // softsign
  r += aw[14] * sp;                                                       // softplus
  r += aw[15] * th;                                                       // tanh
  r += aw[16] * sig;                                                      // sigmoid
  r += aw[17] * (r6 * (1.f / 6.f));                                       // hard_sigmoid
  r += aw[18] * (x * sig);                                                // silu
  r += aw[19] * (x * tanhf(sp));                                          // mish
  return r;
}

// ---------------------------------------------------------------------------
// Softmax of alpha [T,20] and beta [T,2] (tiny)
// ---------------------------------------------------------------------------
__global__ void k_softmax(const float* __restrict__ alpha,
                          const float* __restrict__ beta,
                          float* __restrict__ AW, float* __restrict__ BW) {
  int i = threadIdx.x;
  if (i < T_ITERS) {
    float m = -1e30f;
    for (int k = 0; k < 20; ++k) m = fmaxf(m, alpha[i * 20 + k]);
    float e[20], s = 0.f;
    for (int k = 0; k < 20; ++k) { e[k] = __expf(alpha[i * 20 + k] - m); s += e[k]; }
    float inv = 1.f / s;
    for (int k = 0; k < 20; ++k) AW[i * 20 + k] = e[k] * inv;
    float b0 = beta[i * 2 + 0], b1 = beta[i * 2 + 1];
    float mm = fmaxf(b0, b1);
    float e0 = __expf(b0 - mm), e1 = __expf(b1 - mm);
    float bi = 1.f / (e0 + e1);
    BW[i * 2 + 0] = e0 * bi;
    BW[i * 2 + 1] = e1 * bi;
  }
}

// ---------------------------------------------------------------------------
// W (f32 [N,H]) -> bf16 [N,H] and bf16 transposed [H,N]
// ---------------------------------------------------------------------------
__global__ void k_cvt(const float* __restrict__ W,
                      __bf16* __restrict__ Wbf, __bf16* __restrict__ Wtbf) {
  int idx = blockIdx.x * blockDim.x + threadIdx.x;   // 0 .. N*H-1
  int n = idx / H_DIM, h = idx % H_DIM;
  float w = W[idx];
  Wbf[idx] = f2bf(w);
  Wtbf[(size_t)h * N_DIM + n] = f2bf(w);
}

// ---------------------------------------------------------------------------
// Power iteration: L = ||W||_2^2 (largest eigenvalue of W^T W). One block.
// ---------------------------------------------------------------------------
__global__ void k_power(const float* __restrict__ W, float* __restrict__ scal) {
  __shared__ float v[H_DIM];
  __shared__ float u[N_DIM];
  __shared__ float red[32];
  int t = threadIdx.x;                                // 1024 threads
  v[t] = 1.f;
  __syncthreads();
  float lambda = 1.f;
  for (int it = 0; it < 64; ++it) {
    if (t < N_DIM) {
      float s = 0.f;
      for (int h = 0; h < H_DIM; ++h) s += W[(size_t)t * H_DIM + h] * v[h];
      u[t] = s;
    }
    __syncthreads();
    float s = 0.f;
    for (int n = 0; n < N_DIM; ++n) s += W[(size_t)n * H_DIM + t] * u[n];
    float sq = s * s;
    for (int off = 16; off; off >>= 1) sq += __shfl_down(sq, off, 32);
    if ((t & 31) == 0) red[t >> 5] = sq;
    __syncthreads();
    if (t < 32) {
      float q = red[t];
      for (int off = 16; off; off >>= 1) q += __shfl_down(q, off, 32);
      if (t == 0) red[0] = sqrtf(q);
    }
    __syncthreads();
    lambda = red[0];
    v[t] = s / lambda;
    __syncthreads();
  }
  if (t == 0) {
    scal[0] = lambda;
    scal[1] = 1.f / lambda;
    scal[2] = 0.001f / lambda;
  }
}

// ---------------------------------------------------------------------------
// Zero-init z (in d_out) and z_prev
// ---------------------------------------------------------------------------
__global__ void k_zero(float* __restrict__ Z, float* __restrict__ ZP) {
  size_t i = ((size_t)blockIdx.x * blockDim.x + threadIdx.x) * 4;
  float4 z4 = make_float4(0.f, 0.f, 0.f, 0.f);
  *reinterpret_cast<float4*>(Z + i)  = z4;
  *reinterpret_cast<float4*>(ZP + i) = z4;
}

// ---------------------------------------------------------------------------
// z_aux = z + mom*(z - z_prev), stored bf16 for GEMM1's A operand
// ---------------------------------------------------------------------------
__global__ void k_aux(const float* __restrict__ Z, const float* __restrict__ ZP,
                      __bf16* __restrict__ ZX, float mom) {
  size_t i = ((size_t)blockIdx.x * blockDim.x + threadIdx.x) * 4;
  float4 z4 = *reinterpret_cast<const float4*>(Z + i);
  float4 p4 = *reinterpret_cast<const float4*>(ZP + i);
  v4bf o;
  o[0] = f2bf(z4.x + mom * (z4.x - p4.x));
  o[1] = f2bf(z4.y + mom * (z4.y - p4.y));
  o[2] = f2bf(z4.z + mom * (z4.z - p4.z));
  o[3] = f2bf(z4.w + mom * (z4.w - p4.w));
  *reinterpret_cast<v4bf*>(ZX + i) = o;
}

// ---------------------------------------------------------------------------
// GEMM1: R = bf16( z_aux[B,H] @ W^T  -  x )   -> R bf16 [B,N]
// A = ZX [B, K=1024] row-major bf16; Bt = Wbf [N, K=1024] row-major bf16.
// Block: 256 thr = 8 waves (4x2), wave tile 32x64 -> block tile 128x128.
// ---------------------------------------------------------------------------
__global__ __launch_bounds__(256)
void k_gemm1(const __bf16* __restrict__ A, const __bf16* __restrict__ Bt,
             const float* __restrict__ x, __bf16* __restrict__ R) {
  constexpr int K  = H_DIM;   // 1024
  constexpr int NC = N_DIM;   // 512
  const int lane = threadIdx.x & 31;
  const int wave = threadIdx.x >> 5;
  const int wm = wave >> 1;            // 0..3
  const int wn = wave & 1;             // 0..1
  const int l15 = lane & 15;
  const int lhi = lane >> 4;
  const int mBase = blockIdx.y * 128 + wm * 32;
  const int nBase = blockIdx.x * 128 + wn * 64;

  v8f vzero = {};
  v8f acc[2][4];
#pragma unroll
  for (int i = 0; i < 2; ++i)
#pragma unroll
    for (int j = 0; j < 4; ++j) acc[i][j] = vzero;

  const __bf16* aPtr0 = A + (size_t)(mBase + l15) * K + (lhi << 4);
  const __bf16* aPtr1 = aPtr0 + (size_t)16 * K;
  const __bf16* bPtr[4];
#pragma unroll
  for (int tn = 0; tn < 4; ++tn)
    bPtr[tn] = Bt + (size_t)(nBase + tn * 16 + l15) * K + (lhi << 4);

  for (int k0 = 0; k0 < K; k0 += 32) {
    v16bf a0 = *reinterpret_cast<const v16bf*>(aPtr0 + k0);
    v16bf a1 = *reinterpret_cast<const v16bf*>(aPtr1 + k0);
#pragma unroll
    for (int tn = 0; tn < 4; ++tn) {
      v16bf b = *reinterpret_cast<const v16bf*>(bPtr[tn] + k0);
      acc[0][tn] = wmma_bf16(a0, b, acc[0][tn]);
      acc[1][tn] = wmma_bf16(a1, b, acc[1][tn]);
    }
  }

  // Epilogue: subtract x, store bf16 (becomes GEMM2's A operand)
#pragma unroll
  for (int tm = 0; tm < 2; ++tm)
#pragma unroll
    for (int tn = 0; tn < 4; ++tn)
#pragma unroll
      for (int j = 0; j < 8; ++j) {
        int m = mBase + tm * 16 + (lhi << 3) + j;
        int n = nBase + tn * 16 + l15;
        size_t idx = (size_t)m * NC + n;
        R[idx] = f2bf(acc[tm][tn][j] - x[idx]);
      }
}

// ---------------------------------------------------------------------------
// GEMM2: grad = R[B,N] @ W[N,H]; fused FISTA + 20-activation epilogue.
// A = R [B, K=512] bf16; Bt = Wtbf [H, K=512] bf16 (W transposed).
// Block: 256 thr = 8 waves (2x4), wave tile 32x32 -> block tile 64x128.
// ---------------------------------------------------------------------------
__global__ __launch_bounds__(256)
void k_gemm2(const __bf16* __restrict__ A, const __bf16* __restrict__ Bt,
             float* __restrict__ Z, float* __restrict__ ZP,
             const float* __restrict__ AW, const float* __restrict__ BW,
             const float* __restrict__ scal, int iter, float mom) {
  constexpr int K = N_DIM;    // 512
  const int lane = threadIdx.x & 31;
  const int wave = threadIdx.x >> 5;
  const int wm = wave & 1;             // 0..1
  const int wn = wave >> 1;            // 0..3
  const int l15 = lane & 15;
  const int lhi = lane >> 4;
  const int mBase = blockIdx.y * 64 + wm * 32;
  const int hBase = blockIdx.x * 128 + wn * 32;

  v8f vzero = {};
  v8f acc[2][2];
#pragma unroll
  for (int i = 0; i < 2; ++i)
#pragma unroll
    for (int j = 0; j < 2; ++j) acc[i][j] = vzero;

  const __bf16* aPtr0 = A + (size_t)(mBase + l15) * K + (lhi << 4);
  const __bf16* aPtr1 = aPtr0 + (size_t)16 * K;
  const __bf16* bPtr0 = Bt + (size_t)(hBase + l15) * K + (lhi << 4);
  const __bf16* bPtr1 = bPtr0 + (size_t)16 * K;

  for (int k0 = 0; k0 < K; k0 += 32) {
    v16bf a0 = *reinterpret_cast<const v16bf*>(aPtr0 + k0);
    v16bf a1 = *reinterpret_cast<const v16bf*>(aPtr1 + k0);
    v16bf b0 = *reinterpret_cast<const v16bf*>(bPtr0 + k0);
    v16bf b1 = *reinterpret_cast<const v16bf*>(bPtr1 + k0);
    acc[0][0] = wmma_bf16(a0, b0, acc[0][0]);
    acc[0][1] = wmma_bf16(a0, b1, acc[0][1]);
    acc[1][0] = wmma_bf16(a1, b0, acc[1][0]);
    acc[1][1] = wmma_bf16(a1, b1, acc[1][1]);
  }

  // FISTA + activation-mixture epilogue (pointwise in (b,h): in-place safe)
  const float invL = scal[1];
  const float lam  = scal[2];
  float aw[20];
#pragma unroll
  for (int k = 0; k < 20; ++k) aw[k] = AW[iter * 20 + k];
  const float bw0 = BW[iter * 2 + 0];
  const float bw1 = BW[iter * 2 + 1];

#pragma unroll
  for (int tm = 0; tm < 2; ++tm)
#pragma unroll
    for (int tn = 0; tn < 2; ++tn)
#pragma unroll
      for (int j = 0; j < 8; ++j) {
        int m = mBase + tm * 16 + (lhi << 3) + j;
        int h = hBase + tn * 16 + l15;
        size_t idx = (size_t)m * H_DIM + h;
        float zo  = Z[idx];
        float zpo = ZP[idx];
        float za  = zo + mom * (zo - zpo);
        float zg  = za - acc[tm][tn][j] * invL;
        float zop = mix20(zg, aw, lam);
        Z[idx]  = zop;                     // z      <- z_op
        ZP[idx] = zo * bw0 + zop * bw1;    // z_prev <- z*b0 + z_op*b1
      }
}

// ---------------------------------------------------------------------------
// Host launcher
// ---------------------------------------------------------------------------
extern "C" void kernel_launch(void* const* d_in, const int* in_sizes, int n_in,
                              void* d_out, int out_size, void* d_ws, size_t ws_size,
                              hipStream_t stream) {
  const float* x     = (const float*)d_in[0];   // [B,N,1]
  const float* W     = (const float*)d_in[1];   // [1,N,H]
  const float* alpha = (const float*)d_in[2];   // [T,20]
  const float* beta  = (const float*)d_in[3];   // [T,2]
  float* Z = (float*)d_out;                     // z lives in d_out [B,H,1]

  char* ws = (char*)d_ws;
  float*  scal = (float*)(ws + OFF_SCAL);
  float*  AW   = (float*)(ws + OFF_AW);
  float*  BW   = (float*)(ws + OFF_BW);
  __bf16* Wbf  = (__bf16*)(ws + OFF_WBF);
  __bf16* Wtbf = (__bf16*)(ws + OFF_WTBF);
  __bf16* ZX   = (__bf16*)(ws + OFF_ZAUX);
  __bf16* Rbf  = (__bf16*)(ws + OFF_RBF);
  float*  ZP   = (float*)(ws + OFF_ZP);

  k_softmax<<<1, 32, 0, stream>>>(alpha, beta, AW, BW);
  k_cvt<<<(N_DIM * H_DIM) / 256, 256, 0, stream>>>(W, Wbf, Wtbf);
  k_power<<<1, 1024, 0, stream>>>(W, scal);
  k_zero<<<(B_DIM * H_DIM) / 1024, 256, 0, stream>>>(Z, ZP);

  for (int i = 0; i < T_ITERS; ++i) {
    float mom = (float)i / (float)(i + 3);
    k_aux<<<(B_DIM * H_DIM) / 1024, 256, 0, stream>>>(Z, ZP, ZX, mom);
    k_gemm1<<<dim3(N_DIM / 128, B_DIM / 128), 256, 0, stream>>>(ZX, Wbf, x, Rbf);
    k_gemm2<<<dim3(H_DIM / 128, B_DIM / 64), 256, 0, stream>>>(Rbf, Wtbf, Z, ZP,
                                                               AW, BW, scal, i, mom);
  }
}
